// PriorTransitionModel_84207128805703
// MI455X (gfx1250) — matrix operand order repaired and checked by reading, not compiled, not run on hardware
//
#include <hip/hip_runtime.h>
#include <hip/hip_bf16.h>
#include <math.h>

#define T_STEPS 1024
#define ADIM    64
#define HDIM    2048
#define H3      6144

typedef __bf16 bf16_t;
typedef bf16_t v16bf __attribute__((ext_vector_type(16)));
typedef bf16_t v8bf  __attribute__((ext_vector_type(8)));
typedef float  v8f   __attribute__((ext_vector_type(8)));

// ---------------- device helpers ----------------
__device__ __forceinline__ float bf_lo(unsigned u) { return __uint_as_float(u << 16); }
__device__ __forceinline__ float bf_hi(unsigned u) { return __uint_as_float(u & 0xffff0000u); }
__device__ __forceinline__ float sigmoidf_(float x) { return 1.0f / (1.0f + expf(-x)); }
__device__ __forceinline__ float eluf_(float x)     { return x > 0.0f ? x : expm1f(x); }
__device__ __forceinline__ float softplusf_(float x){ return x > 20.0f ? x : log1pf(expf(x)); }

// accumulate 8 bf16 weights (one uint4) against 8 staged f32 activations
__device__ __forceinline__ void dot8(float& acc, uint4 u, const float4& sa, const float4& sb) {
  acc += bf_lo(u.x) * sa.x; acc += bf_hi(u.x) * sa.y;
  acc += bf_lo(u.y) * sa.z; acc += bf_hi(u.y) * sa.w;
  acc += bf_lo(u.z) * sb.x; acc += bf_hi(u.z) * sb.y;
  acc += bf_lo(u.w) * sb.z; acc += bf_hi(u.w) * sb.w;
}

// A-fragment (16x32 bf16, row-major source): lane holds row M=lrow.
// Contiguous K run of 16 at k0 + 16*hi (matches B fragment order).
__device__ __forceinline__ v16bf load_a_frag(const bf16_t* __restrict__ row, int k0, int hi) {
  return *(const v16bf*)(row + k0 + 16 * hi);
}

// B-fragment (32x16 bf16) for B = W^T, W row-major: lane holds column N=lrow
// (== row n of W), contiguous K run of 16 at k0 + 16*hi.
__device__ __forceinline__ v16bf load_b_frag(const bf16_t* __restrict__ wrow, int k0, int hi) {
  return *(const v16bf*)(wrow + k0 + 16 * hi);
}

// ---------------- conversion ----------------
__global__ void cvt_f32_bf16(const float* __restrict__ src, __hip_bfloat16* __restrict__ dst, int n) {
  int i = blockIdx.x * blockDim.x + threadIdx.x;
  int stride = gridDim.x * blockDim.x;
  for (; i < n; i += stride) dst[i] = __float2bfloat16(src[i]);
}

__global__ void init_sync_kernel(unsigned* sync) {
  if (threadIdx.x < 2) sync[threadIdx.x] = 0u;
}

// ---------------- GI = actions @ W_ih^T + b_ih  (WMMA bf16) ----------------
__global__ void __launch_bounds__(256)
gi_kernel(const bf16_t* __restrict__ Abf, const bf16_t* __restrict__ Wbf,
          const float* __restrict__ bih, float* __restrict__ GI) {
  const int wave = threadIdx.x >> 5, lane = threadIdx.x & 31;
  const int lrow = lane & 15, hi = lane >> 4;
  const int NTILE = H3 / 16;                       // 384
  int tile = blockIdx.x * (blockDim.x >> 5) + wave;
  if (tile >= (T_STEPS / 16) * NTILE) return;
  int mt = tile / NTILE, nt = tile - mt * NTILE;

  const bf16_t* arow = Abf + (size_t)(mt * 16 + lrow) * ADIM;
  const bf16_t* wrow = Wbf + (size_t)(nt * 16 + lrow) * ADIM;
  v8f acc = {};
#pragma unroll
  for (int k0 = 0; k0 < ADIM; k0 += 32) {
    v16bf a = load_a_frag(arow, k0, hi);
    v16bf b = load_b_frag(wrow, k0, hi);
    acc = __builtin_amdgcn_wmma_f32_16x16x32_bf16(false, a, false, b, (short)0, acc, false, false);
  }
  int n = nt * 16 + lrow;
  float bias = bih[n];
#pragma unroll
  for (int r = 0; r < 8; ++r) {
    int m = mt * 16 + r + 8 * hi;
    GI[(size_t)m * H3 + n] = acc[r] + bias;
  }
}

// ---------------- persistent sequential core ----------------
#define NBLK 128
#define NTHR 256
#define NWAVE (NBLK * (NTHR / 32))   // 1024 waves; H=2048 -> 2 outputs/wave/phase

__device__ __forceinline__ void grid_sync(unsigned* cnt, unsigned* gen) {
  __threadfence();
  __syncthreads();
  if (threadIdx.x == 0) {
    unsigned g = __atomic_load_n(gen, __ATOMIC_RELAXED);
    if (atomicAdd(cnt, 1u) == (unsigned)(NBLK - 1)) {
      __atomic_store_n(cnt, 0u, __ATOMIC_RELAXED);
      __atomic_fetch_add(gen, 1u, __ATOMIC_RELEASE);
    } else {
      while (__atomic_load_n(gen, __ATOMIC_ACQUIRE) == g) __builtin_amdgcn_s_sleep(1);
    }
  }
  __syncthreads();
  __threadfence();
}

__global__ void __launch_bounds__(NTHR)
seq_kernel(const float* __restrict__ GI,
           const uint4* __restrict__ Whh,   // bf16 x8 packs, [3H][H/8]
           const float* __restrict__ bhh,
           const uint4* __restrict__ W1, const float* __restrict__ b1,
           const uint4* __restrict__ W2, const float* __restrict__ b2,
           const float* __restrict__ h0,
           float* __restrict__ S, __hip_bfloat16* __restrict__ Sbf,
           float* __restrict__ abuf, float* __restrict__ bbuf,
           unsigned* __restrict__ sync) {
  __shared__ float sv[HDIM];                      // 8 KB: staged activation vector
  const int wave = threadIdx.x >> 5, lane = threadIdx.x & 31;
  const int gw = blockIdx.x * (NTHR / 32) + wave;
  const int HW4 = HDIM / 8;                       // uint4 packs per weight row (256)

  for (int t = 0; t < T_STEPS; ++t) {
    const float* sp = (t == 0) ? h0 : (S + (size_t)(t - 1) * HDIM);
    for (int i = threadIdx.x; i < HDIM; i += NTHR) sv[i] = sp[i];
    __syncthreads();
    const float* gi = GI + (size_t)t * H3;

    // Phase A: gh = s_prev @ W_hh^T, gates, h_new, elu -> abuf
    for (int i = gw; i < HDIM; i += NWAVE) {
      const uint4* r0 = Whh + (size_t)i * HW4;
      const uint4* r1 = Whh + (size_t)(HDIM + i) * HW4;
      const uint4* r2 = Whh + (size_t)(2 * HDIM + i) * HW4;
      float dr = 0.f, dz = 0.f, dn = 0.f;
#pragma unroll 2
      for (int k = lane; k < HW4; k += 32) {
        const float4* s4 = (const float4*)&sv[8 * k];
        float4 sa = s4[0], sb = s4[1];
        dot8(dr, r0[k], sa, sb);
        dot8(dz, r1[k], sa, sb);
        dot8(dn, r2[k], sa, sb);
      }
#pragma unroll
      for (int off = 16; off; off >>= 1) {
        dr += __shfl_xor(dr, off, 32);
        dz += __shfl_xor(dz, off, 32);
        dn += __shfl_xor(dn, off, 32);
      }
      if (lane == 0) {
        float r = sigmoidf_(gi[i] + bhh[i] + dr);
        float z = sigmoidf_(gi[HDIM + i] + bhh[HDIM + i] + dz);
        float nn = tanhf(gi[2 * HDIM + i] + r * (bhh[2 * HDIM + i] + dn));
        float hn = (1.f - z) * nn + z * sv[i];
        abuf[i] = eluf_(hn);
      }
    }
    grid_sync(&sync[0], &sync[1]);

    // Phase B: b = elu(a @ fc1^T + b1)
    for (int i = threadIdx.x; i < HDIM; i += NTHR) sv[i] = abuf[i];
    __syncthreads();
    for (int j = gw; j < HDIM; j += NWAVE) {
      const uint4* w = W1 + (size_t)j * HW4;
      float d = 0.f;
#pragma unroll 2
      for (int k = lane; k < HW4; k += 32) {
        const float4* s4 = (const float4*)&sv[8 * k];
        dot8(d, w[k], s4[0], s4[1]);
      }
#pragma unroll
      for (int off = 16; off; off >>= 1) d += __shfl_xor(d, off, 32);
      if (lane == 0) bbuf[j] = eluf_(d + b1[j]);
    }
    grid_sync(&sync[0], &sync[1]);

    // Phase C: s = elu(b @ fc2^T + b2) -> S[t] (f32) and Sbf[t] (bf16)
    for (int i = threadIdx.x; i < HDIM; i += NTHR) sv[i] = bbuf[i];
    __syncthreads();
    for (int j = gw; j < HDIM; j += NWAVE) {
      const uint4* w = W2 + (size_t)j * HW4;
      float d = 0.f;
#pragma unroll 2
      for (int k = lane; k < HW4; k += 32) {
        const float4* s4 = (const float4*)&sv[8 * k];
        dot8(d, w[k], s4[0], s4[1]);
      }
#pragma unroll
      for (int off = 16; off; off >>= 1) d += __shfl_xor(d, off, 32);
      if (lane == 0) {
        float v = eluf_(d + b2[j]);
        S[(size_t)t * HDIM + j] = v;
        Sbf[(size_t)t * HDIM + j] = __float2bfloat16(v);
      }
    }
    grid_sync(&sync[0], &sync[1]);
  }
}

// ---------------- deferred heads: means / softplus stds (WMMA bf16) ----------------
__global__ void __launch_bounds__(256)
head_kernel(const bf16_t* __restrict__ Sbf,
            const bf16_t* __restrict__ Wm, const float* __restrict__ bm,
            const bf16_t* __restrict__ Ws, const float* __restrict__ bs,
            float* __restrict__ out) {
  const int wave = threadIdx.x >> 5, lane = threadIdx.x & 31;
  const int lrow = lane & 15, hi = lane >> 4;
  int tile = blockIdx.x * (blockDim.x >> 5) + wave;   // 2 * 64 * 128 = 16384 tiles
  if (tile >= 2 * (T_STEPS / 16) * (HDIM / 16)) return;
  int which = tile >> 13;            // 8192 tiles per head
  int t2 = tile & 8191;
  int mt = t2 >> 7;                  // 128 N-tiles
  int nt = t2 & 127;

  const bf16_t* W = which ? Ws : Wm;
  const float* bias = which ? bs : bm;
  const bf16_t* arow = Sbf + (size_t)(mt * 16 + lrow) * HDIM;
  const bf16_t* wrow = W + (size_t)(nt * 16 + lrow) * HDIM;

  v8f acc = {};
  for (int k0 = 0; k0 < HDIM; k0 += 32) {
    v16bf a = load_a_frag(arow, k0, hi);
    v16bf b = load_b_frag(wrow, k0, hi);
    acc = __builtin_amdgcn_wmma_f32_16x16x32_bf16(false, a, false, b, (short)0, acc, false, false);
  }
  int n = nt * 16 + lrow;
  float bn = bias[n];
  float* obase = out + (size_t)which * (size_t)T_STEPS * HDIM;
#pragma unroll
  for (int r = 0; r < 8; ++r) {
    int m = mt * 16 + r + 8 * hi;
    float v = acc[r] + bn;
    if (which) v = softplusf_(v);
    obase[(size_t)m * HDIM + n] = v;
  }
}

// ---------------- host launch ----------------
extern "C" void kernel_launch(void* const* d_in, const int* in_sizes, int n_in,
                              void* d_out, int out_size, void* d_ws, size_t ws_size,
                              hipStream_t stream) {
  (void)in_sizes; (void)n_in; (void)out_size; (void)ws_size;
  const float* actions = (const float*)d_in[0];
  const float* state   = (const float*)d_in[1];
  const float* W_ih    = (const float*)d_in[2];
  const float* W_hh    = (const float*)d_in[3];
  const float* b_ih    = (const float*)d_in[4];
  const float* b_hh    = (const float*)d_in[5];
  const float* fc1_w   = (const float*)d_in[6];
  const float* fc1_b   = (const float*)d_in[7];
  const float* fc2_w   = (const float*)d_in[8];
  const float* fc2_b   = (const float*)d_in[9];
  const float* mean_w  = (const float*)d_in[10];
  const float* mean_b  = (const float*)d_in[11];
  const float* std_w   = (const float*)d_in[12];
  const float* std_b   = (const float*)d_in[13];

  char* ws = (char*)d_ws;
  size_t off = 0;
  auto take = [&](size_t bytes) -> void* {
    void* p = ws + off;
    off += (bytes + 255) & ~(size_t)255;
    return p;
  };
  float*          GI    = (float*)take(sizeof(float) * T_STEPS * H3);      // 25.2 MB
  float*          S     = (float*)take(sizeof(float) * T_STEPS * HDIM);    //  8.4 MB
  __hip_bfloat16* Sbf   = (__hip_bfloat16*)take(2ull * T_STEPS * HDIM);    //  4.2 MB
  __hip_bfloat16* Abf   = (__hip_bfloat16*)take(2ull * T_STEPS * ADIM);
  __hip_bfloat16* Wihb  = (__hip_bfloat16*)take(2ull * H3 * ADIM);
  __hip_bfloat16* Whhb  = (__hip_bfloat16*)take(2ull * H3 * HDIM);         // 25.2 MB
  __hip_bfloat16* W1b   = (__hip_bfloat16*)take(2ull * HDIM * HDIM);       //  8.4 MB
  __hip_bfloat16* W2b   = (__hip_bfloat16*)take(2ull * HDIM * HDIM);
  __hip_bfloat16* Wmb   = (__hip_bfloat16*)take(2ull * HDIM * HDIM);
  __hip_bfloat16* Wsb   = (__hip_bfloat16*)take(2ull * HDIM * HDIM);
  float*          abuf  = (float*)take(sizeof(float) * HDIM);
  float*          bbuf  = (float*)take(sizeof(float) * HDIM);
  unsigned*       syncb = (unsigned*)take(256);

  const int CB = 1024, CT = 256;
  cvt_f32_bf16<<<CB, CT, 0, stream>>>(actions, Abf, T_STEPS * ADIM);
  cvt_f32_bf16<<<CB, CT, 0, stream>>>(W_ih,   Wihb, H3 * ADIM);
  cvt_f32_bf16<<<CB, CT, 0, stream>>>(W_hh,   Whhb, H3 * HDIM);
  cvt_f32_bf16<<<CB, CT, 0, stream>>>(fc1_w,  W1b,  HDIM * HDIM);
  cvt_f32_bf16<<<CB, CT, 0, stream>>>(fc2_w,  W2b,  HDIM * HDIM);
  cvt_f32_bf16<<<CB, CT, 0, stream>>>(mean_w, Wmb,  HDIM * HDIM);
  cvt_f32_bf16<<<CB, CT, 0, stream>>>(std_w,  Wsb,  HDIM * HDIM);
  init_sync_kernel<<<1, 64, 0, stream>>>(syncb);

  // GI GEMM: 24576 16x16 tiles, 8 waves/block
  gi_kernel<<<3072, 256, 0, stream>>>((const bf16_t*)Abf, (const bf16_t*)Wihb, b_ih, GI);

  // Persistent sequential recurrence
  seq_kernel<<<NBLK, NTHR, 0, stream>>>(GI,
                                        (const uint4*)Whhb, b_hh,
                                        (const uint4*)W1b, fc1_b,
                                        (const uint4*)W2b, fc2_b,
                                        state, S, Sbf, abuf, bbuf, syncb);

  // Deferred heads: 16384 tiles, 8 waves/block
  head_kernel<<<2048, 256, 0, stream>>>((const bf16_t*)Sbf,
                                        (const bf16_t*)Wmb, mean_b,
                                        (const bf16_t*)Wsb, std_b,
                                        (float*)d_out);
}
